// Contrastiveloss_14809047236869
// MI455X (gfx1250) — matrix operand order repaired and checked by reading, compiled, not run on hardware
//
#include <hip/hip_runtime.h>

// ---------------------------------------------------------------------------
// Contrastive (SupCon) loss, fused:  loss = mean_i( logsumexp_j s_ij -
//                                                   logsumexp_{y_j==y_i} s_ij )
// s = (Xn Xn^T)/tau, Xn = row-normalized x.  N=8192, D=2048, tau=0.1.
// |s| <= 10 => exp(s) <= 2.2e4, row sums <= 1.8e8: fp32-safe without max
// subtraction, so plain exp-sum accumulation replaces online softmax.
// 275 GFLOP GEMM, operands (32MB f16) fit the 192MB L2 -> compute bound ->
// v_wmma_f32_16x16x32_f16 path with async global->LDS double buffering.
// ---------------------------------------------------------------------------

typedef _Float16 half4  __attribute__((ext_vector_type(4)));
typedef _Float16 half8  __attribute__((ext_vector_type(8)));
typedef _Float16 half16 __attribute__((ext_vector_type(16)));
typedef float    v8f    __attribute__((ext_vector_type(8)));
typedef int      b128_t __attribute__((vector_size(16)));   // matches builtin param

#define N_ROWS  8192
#define DIM     2048
#define TAU_INV 10.0f

#define BM 128          // rows of S per block
#define BN 128          // cols of S per block
#define BK 32           // K-step (matches v_wmma_f32_16x16x32_f16)
#define LDH (BK + 8)    // LDS row stride in halfs: 80B (16B-aligned, bank-shifted)

// ----- CDNA5 async global->LDS copy (ASYNCcnt-tracked), with safe fallback --
#if defined(__has_builtin)
#if __has_builtin(__builtin_amdgcn_global_load_async_to_lds_b128) && \
    __has_builtin(__builtin_amdgcn_s_wait_asynccnt)
#define USE_ASYNC_LDS 1
#endif
#endif

#if defined(USE_ASYNC_LDS)
// Param types per hipcc diagnostic: (int4 AS1*, int4 AS3*, imm offset, imm cpol)
#define ASYNC_CP16(g, l)                                                    \
  __builtin_amdgcn_global_load_async_to_lds_b128(                           \
      (__attribute__((address_space(1))) b128_t*)(void*)(g),                \
      (__attribute__((address_space(3))) b128_t*)(void*)(l), 0, 0)
#define WAIT_ASYNC() __builtin_amdgcn_s_wait_asynccnt(0)
#else
#define ASYNC_CP16(g, l) (*(half8*)(l) = *(const half8*)(g))
#define WAIT_ASYNC() ((void)0)
#endif

// ---------------------------------------------------------------------------
// Kernel 1: row L2-normalize, f32 -> f16
// ---------------------------------------------------------------------------
__global__ __launch_bounds__(256) void normalize_rows(const float* __restrict__ x,
                                                      _Float16* __restrict__ xn) {
  const int row = blockIdx.x;
  const int tid = threadIdx.x;
  const float4* xr = (const float4*)(x + (size_t)row * DIM);
  const float4 a = xr[tid];          // DIM/4 = 512 float4 per row
  const float4 b = xr[tid + 256];
  float ss = a.x * a.x + a.y * a.y + a.z * a.z + a.w * a.w
           + b.x * b.x + b.y * b.y + b.z * b.z + b.w * b.w;
#pragma unroll
  for (int off = 16; off >= 1; off >>= 1) ss += __shfl_xor(ss, off, 32);
  __shared__ float wss[8];
  const int wave = tid >> 5, lane = tid & 31;
  if (lane == 0) wss[wave] = ss;
  __syncthreads();
  float tot = 0.f;
#pragma unroll
  for (int i = 0; i < 8; ++i) tot += wss[i];
  const float rn = rsqrtf(tot);

  half4 ha, hb;
  ha.x = (_Float16)(a.x * rn); ha.y = (_Float16)(a.y * rn);
  ha.z = (_Float16)(a.z * rn); ha.w = (_Float16)(a.w * rn);
  hb.x = (_Float16)(b.x * rn); hb.y = (_Float16)(b.y * rn);
  hb.z = (_Float16)(b.z * rn); hb.w = (_Float16)(b.w * rn);
  _Float16* o = xn + (size_t)row * DIM;
  *(half4*)(o + (size_t)tid * 4)         = ha;
  *(half4*)(o + (size_t)(tid + 256) * 4) = hb;
}

// ---------------------------------------------------------------------------
// Kernel 2: tiled WMMA GEMM of a 128x128 S tile + fused exp / masked row sums.
// 256 threads = 8 waves arranged 2(M) x 4(N); wave tile 64x32 = 4x2 WMMA
// (8 v_wmma per K-step per wave). Double-buffered LDS, async fills.
// ---------------------------------------------------------------------------
__global__ __launch_bounds__(256) void gemm_exp_sums(const _Float16* __restrict__ xn,
                                                     const int* __restrict__ y,
                                                     float* __restrict__ gAll,
                                                     float* __restrict__ gSame) {
  __shared__ _Float16 As[2][BM][LDH];
  __shared__ _Float16 Bs[2][BN][LDH];
  __shared__ float rowAll[BM];
  __shared__ float rowSame[BM];
  __shared__ int   yRow[BM];
  __shared__ int   yCol[BN];

  const int tid     = threadIdx.x;
  const int lane    = tid & 31;
  const int wave    = tid >> 5;
  const int waveRow = wave >> 2;  // 0..1  -> 64 rows each
  const int waveCol = wave & 3;   // 0..3  -> 32 cols each
  const int lrow    = lane & 15;  // position within 16-lane half
  const int lgrp    = lane >> 4;  // 0 or 1
  const int rowBase = blockIdx.y * BM;
  const int colBase = blockIdx.x * BN;

  if (tid < BM) {
    rowAll[tid] = 0.f; rowSame[tid] = 0.f;
    yRow[tid] = y[rowBase + tid];
    yCol[tid] = y[colBase + tid];
  }

  // Branch-free cooperative fill: (128+128) rows x 32 halfs = 1024 x 16B
  // chunks, 4 per thread; `it` is compile-time so the As/Bs split costs no
  // exec-mask divergence.  it=0,1 -> As rows 0..127; it=2,3 -> Bs rows 0..127.
  auto fill = [&](int b, int kk) {
#pragma unroll
    for (int it = 0; it < 4; ++it) {
      const int c  = it * 256 + tid;
      const int r  = c >> 2;          // 0..255
      const int cc = (c & 3) * 8;     // 0,8,16,24 halfs
      if (it < 2) {
        ASYNC_CP16(xn + (size_t)(rowBase + r) * DIM + kk + cc, &As[b][r][cc]);
      } else {
        ASYNC_CP16(xn + (size_t)(colBase + r - BM) * DIM + kk + cc,
                   &Bs[b][r - BM][cc]);
      }
    }
  };

  v8f acc[4][2] = {};

  fill(0, 0);  // prologue: buffer 0 in flight

  for (int k0 = 0; k0 < DIM; k0 += BK) {
    const int buf = (k0 >> 5) & 1;
    WAIT_ASYNC();      // this wave's fills for As/Bs[buf] are complete
    __syncthreads();   // all waves' fills visible; prev compute (buf^1) retired
    if (k0 + BK < DIM) fill(buf ^ 1, k0 + BK);  // DMA overlaps the WMMAs below

    // A fragment (16x32 f16, ISA 7.12.2): lane m=L%16; K octets at 8*(L/16), +16.
    half16 afrag[4], bfrag[2];
#pragma unroll
    for (int mt = 0; mt < 4; ++mt) {
      const int m  = waveRow * 64 + mt * 16 + lrow;
      const int ka = lgrp * 8;
      half8 lo = *(const half8*)&As[buf][m][ka];
      half8 hi = *(const half8*)&As[buf][m][ka + 16];
#pragma unroll
      for (int i = 0; i < 8; ++i) { afrag[mt][i] = lo[i]; afrag[mt][i + 8] = hi[i]; }
    }
    // B fragment (32x16): lane n=L%16; K octets at 8*(L/16), +16 (same as A).
#pragma unroll
    for (int nt = 0; nt < 2; ++nt) {
      const int n  = waveCol * 32 + nt * 16 + lrow;
      const int kb = lgrp * 8;
      half8 lo = *(const half8*)&Bs[buf][n][kb];
      half8 hi = *(const half8*)&Bs[buf][n][kb + 16];
#pragma unroll
      for (int i = 0; i < 8; ++i) { bfrag[nt][i] = lo[i]; bfrag[nt][i + 8] = hi[i]; }
    }

#pragma unroll
    for (int mt = 0; mt < 4; ++mt)
#pragma unroll
      for (int nt = 0; nt < 2; ++nt)
        acc[mt][nt] = __builtin_amdgcn_wmma_f32_16x16x32_f16(
            false, afrag[mt], false, bfrag[nt], (short)0, acc[mt][nt], false, false);
  }

  // Epilogue: exp(10*s), per-row (all / same-class) partial sums.
  __syncthreads();
#pragma unroll
  for (int mt = 0; mt < 4; ++mt) {
#pragma unroll
    for (int nt = 0; nt < 2; ++nt) {
      const int n    = waveCol * 32 + nt * 16 + lrow;
      const int ycol = yCol[n];
#pragma unroll
      for (int r = 0; r < 8; ++r) {
        // C/D layout: VGPR r -> M = r (lanes 0-15) or r+8 (lanes 16-31); N = L%16.
        const int m = waveRow * 64 + mt * 16 + lgrp * 8 + r;
        const float e = __expf(acc[mt][nt][r] * TAU_INV);
        float eAll  = e;
        float eSame = (yRow[m] == ycol) ? e : 0.f;
#pragma unroll
        for (int off = 1; off < 16; off <<= 1) {
          eAll  += __shfl_xor(eAll,  off, 32);
          eSame += __shfl_xor(eSame, off, 32);
        }
        if (lrow == 0) {  // lanes 0 and 16 own distinct rows m
          atomicAdd(&rowAll[m],  eAll);
          atomicAdd(&rowSame[m], eSame);
        }
      }
    }
  }
  __syncthreads();
  if (tid < BM) {
    atomicAdd(&gAll[rowBase + tid],  rowAll[tid]);
    atomicAdd(&gSame[rowBase + tid], rowSame[tid]);
  }
}

// ---------------------------------------------------------------------------
// Kernel 3: loss = mean(log(sumAll) - log(sumSame))
// ---------------------------------------------------------------------------
__global__ __launch_bounds__(256) void finalize_loss(const float* __restrict__ gAll,
                                                     const float* __restrict__ gSame,
                                                     float* __restrict__ out) {
  float s = 0.f;
  for (int i = threadIdx.x; i < N_ROWS; i += 256)
    s += __logf(gAll[i]) - __logf(gSame[i]);
#pragma unroll
  for (int off = 16; off >= 1; off >>= 1) s += __shfl_xor(s, off, 32);
  __shared__ float wss[8];
  const int wave = threadIdx.x >> 5, lane = threadIdx.x & 31;
  if (lane == 0) wss[wave] = s;
  __syncthreads();
  if (threadIdx.x == 0) {
    float tot = 0.f;
    for (int i = 0; i < 8; ++i) tot += wss[i];
    out[0] = tot / (float)N_ROWS;
  }
}

__global__ void zero_f32(float* __restrict__ p, int n) {
  const int i = blockIdx.x * blockDim.x + threadIdx.x;
  if (i < n) p[i] = 0.f;
}

// ---------------------------------------------------------------------------
extern "C" void kernel_launch(void* const* d_in, const int* in_sizes, int n_in,
                              void* d_out, int out_size, void* d_ws, size_t ws_size,
                              hipStream_t stream) {
  const float* x = (const float*)d_in[0];
  const int*   y = (const int*)d_in[1];   // labels 0..99 (harness int mapping)
  float* out = (float*)d_out;

  // Workspace layout: [ xn f16 : N*D*2 = 32MB ][ sumAll : N f32 ][ sumSame : N f32 ]
  _Float16* xn    = (_Float16*)d_ws;
  float*    gAll  = (float*)((char*)d_ws + (size_t)N_ROWS * DIM * sizeof(_Float16));
  float*    gSame = gAll + N_ROWS;

  zero_f32<<<(2 * N_ROWS + 255) / 256, 256, 0, stream>>>(gAll, 2 * N_ROWS);
  normalize_rows<<<N_ROWS, 256, 0, stream>>>(x, xn);
  dim3 grid(N_ROWS / BN, N_ROWS / BM);   // (64, 64)
  gemm_exp_sums<<<grid, 256, 0, stream>>>(xn, y, gAll, gSame);
  finalize_loss<<<1, 256, 0, stream>>>(gAll, gSame, out);
}
